// Head_6459630813507
// MI455X (gfx1250) — matrix operand order repaired and checked by reading, not compiled, not run on hardware
//
#include <hip/hip_runtime.h>

#define BATCH 8
#define TSEQ  2048
#define CDIM  768
#define HDIM  64
#define NW    4   // waves per block (wave32)

typedef __attribute__((ext_vector_type(16))) __bf16 v16bf;
typedef __attribute__((ext_vector_type(8)))  float  v8f;

union Frag {
  v16bf v;
  unsigned int u[8];
};

// round-to-nearest-even fp32 -> bf16 (returned in low 16 bits)
static __device__ __forceinline__ unsigned int f2bf(float f) {
  union { float f; unsigned int u; } c; c.f = f;
  unsigned int r = c.u + 0x7FFFu + ((c.u >> 16) & 1u);
  return r >> 16;
}

// ---------------------------------------------------------------------------
// Kernel 0: W [C,H] fp32 -> W^T [H,C] bf16
// ---------------------------------------------------------------------------
__global__ void wconvT_kernel(const float* __restrict__ w,
                              unsigned short* __restrict__ wt) {
  int idx = blockIdx.x * blockDim.x + threadIdx.x;
  if (idx < CDIM * HDIM) {
    int c = idx / HDIM;
    int h = idx - c * HDIM;
    wt[h * CDIM + c] = (unsigned short)f2bf(w[idx]);
  }
}

// ---------------------------------------------------------------------------
// Kernel 1: fused QKV projection.
//   X [B*T, C] fp32  x  W^T [H, C] bf16  ->  Q,K [B*T,H] bf16,  V^T [B][H,T] bf16
//   Q pre-scaled by H^-0.5 = 0.125.
// One wave per 16-row tile; 12 f32 accumulators (Q/K/V x 4 h-tiles).
// ---------------------------------------------------------------------------
__global__ __launch_bounds__(NW * 32) void qkv_proj_kernel(
    const float* __restrict__ x,
    const unsigned int* __restrict__ wqT,
    const unsigned int* __restrict__ wkT,
    const unsigned int* __restrict__ wvT,
    unsigned short* __restrict__ qbf,
    unsigned short* __restrict__ kbf,
    unsigned short* __restrict__ vtbf) {
  const int lane = threadIdx.x & 31;
  const int wv   = threadIdx.x >> 5;
  const int m    = lane & 15;
  const int half = lane >> 4;
  const int r0   = (blockIdx.x * NW + wv) * 16;  // global row in [B*T]

  const v8f vzero = {0.f,0.f,0.f,0.f,0.f,0.f,0.f,0.f};
  v8f acc[12];
#pragma unroll
  for (int i = 0; i < 12; ++i) acc[i] = vzero;

  const int aRow = r0 + m;
#pragma unroll 1
  for (int kc = 0; kc < CDIM / 32; ++kc) {
    // A fragment: 16x32 bf16 chunk of X (fp32 -> bf16, contiguous pairs)
    Frag a;
#pragma unroll
    for (int j = 0; j < 8; ++j) {
      int k0 = 2 * j + (half ? 16 : 0) + kc * 32;
      const float* p = x + aRow * CDIM + k0;
      a.u[j] = f2bf(p[0]) | (f2bf(p[1]) << 16);
    }
    const unsigned int* wt[3] = { wqT, wkT, wvT };
#pragma unroll
    for (int mat = 0; mat < 3; ++mat) {
#pragma unroll
      for (int ht = 0; ht < 4; ++ht) {
        // B fragment 32x16: B[k,n] = W^T[ht*16+n][kc*32+k], pairs contiguous
        Frag b;
        int nrow = ht * 16 + m;
#pragma unroll
        for (int j = 0; j < 8; ++j) {
          int kk = 2 * j + (half ? 16 : 0) + kc * 32;
          b.u[j] = wt[mat][(nrow * CDIM + kk) >> 1];
        }
        acc[mat * 4 + ht] = __builtin_amdgcn_wmma_f32_16x16x32_bf16(
            false, a.v, false, b.v, (short)0, acc[mat * 4 + ht], false, false);
      }
    }
  }

  const int b  = r0 / TSEQ;
  const int tl = r0 - b * TSEQ;
#pragma unroll
  for (int ht = 0; ht < 4; ++ht) {
#pragma unroll
    for (int i = 0; i < 8; ++i) {
      int row = i + 8 * half;          // C-layout: slot i, lane-half -> M
      int h   = ht * 16 + m;           // N = lane%16
      qbf[(r0 + row) * HDIM + h] = (unsigned short)f2bf(acc[ht][i] * 0.125f);
      kbf[(r0 + row) * HDIM + h] = (unsigned short)f2bf(acc[4 + ht][i]);
      vtbf[b * (HDIM * TSEQ) + h * TSEQ + (tl + row)] =
          (unsigned short)f2bf(acc[8 + ht][i]);
    }
  }
}

// ---------------------------------------------------------------------------
// Kernel 2: flash attention, one wave per 16-row query tile, key blocks of 32.
// ---------------------------------------------------------------------------
__global__ __launch_bounds__(NW * 32) void flash_attn_kernel(
    const unsigned int* __restrict__ qbf,
    const unsigned int* __restrict__ kbf,
    const unsigned int* __restrict__ vtbf,
    float* __restrict__ out) {
  __shared__ unsigned int pstage[NW][16 * 32 / 2];  // 16x32 bf16 P per wave

  const int lane = threadIdx.x & 31;
  const int wv   = threadIdx.x >> 5;
  const int m    = lane & 15;
  const int half = lane >> 4;
  const int tile = blockIdx.x * NW + wv;          // 0..B*T/16-1
  const int b    = tile >> 7;                     // 128 tiles per batch
  const int t0   = (tile & 127) * 16;
  const int rowbase = b * TSEQ + t0;

  const unsigned int* kb_b = kbf  + (b * TSEQ * HDIM) / 2;
  const unsigned int* vt_b = vtbf + (b * HDIM * TSEQ) / 2;

  // Q A-fragments (16x32 each, h split 0..31 / 32..63); scale already folded in
  Frag qa[2];
#pragma unroll
  for (int c = 0; c < 2; ++c)
#pragma unroll
    for (int j = 0; j < 8; ++j) {
      int k0 = 2 * j + (half ? 16 : 0) + c * 32;
      qa[c].u[j] = qbf[((rowbase + m) * HDIM + k0) >> 1];
    }

  const v8f vzero = {0.f,0.f,0.f,0.f,0.f,0.f,0.f,0.f};
  v8f o[4];
#pragma unroll
  for (int i = 0; i < 4; ++i) o[i] = vzero;
  float mrun[8], lrun[8];
#pragma unroll
  for (int i = 0; i < 8; ++i) { mrun[i] = -1e30f; lrun[i] = 0.f; }

  const int nblocks = ((t0 + 15) >> 5) + 1;  // key blocks needed (causal)
  const int nfull   = t0 >> 5;               // blocks with no masking

#pragma unroll 1
  for (int kb = 0; kb < nblocks; ++kb) {
    const int kb0 = kb * 32;

    // S = Q * K^T : contract over h (2 chunks), N split into two 16-wide tiles
    v8f s0 = vzero, s1 = vzero;
#pragma unroll
    for (int hc = 0; hc < 2; ++hc) {
      Frag kf0, kf1;  // B[h,n] = K[kb0+n, h], h-pairs contiguous in [T,H]
#pragma unroll
      for (int j = 0; j < 8; ++j) {
        int kk = 2 * j + (half ? 16 : 0) + hc * 32;
        kf0.u[j] = kb_b[((kb0 + m) * HDIM + kk) >> 1];
        kf1.u[j] = kb_b[((kb0 + 16 + m) * HDIM + kk) >> 1];
      }
      s0 = __builtin_amdgcn_wmma_f32_16x16x32_bf16(
          false, qa[hc].v, false, kf0.v, (short)0, s0, false, false);
      s1 = __builtin_amdgcn_wmma_f32_16x16x32_bf16(
          false, qa[hc].v, false, kf1.v, (short)0, s1, false, false);
    }

    if (kb >= nfull) {  // diagonal block: causal mask (uniform branch)
#pragma unroll
      for (int i = 0; i < 8; ++i) {
        int qrow = t0 + i + 8 * half;
        if (kb0 + m      > qrow) s0[i] = -1e30f;
        if (kb0 + 16 + m > qrow) s1[i] = -1e30f;
      }
    }

    // online softmax update; rows live in (slot, lane-half), cols across 16 lanes
    unsigned short* ps = (unsigned short*)pstage[wv];
#pragma unroll
    for (int i = 0; i < 8; ++i) {
      float mx = fmaxf(s0[i], s1[i]);
      mx = fmaxf(mx, __shfl_xor(mx, 1));
      mx = fmaxf(mx, __shfl_xor(mx, 2));
      mx = fmaxf(mx, __shfl_xor(mx, 4));
      mx = fmaxf(mx, __shfl_xor(mx, 8));
      float mn = fmaxf(mrun[i], mx);
      float sc = __expf(mrun[i] - mn);
      mrun[i] = mn;
      float p0 = __expf(s0[i] - mn);
      float p1 = __expf(s1[i] - mn);
      float rs = p0 + p1;
      rs += __shfl_xor(rs, 1);
      rs += __shfl_xor(rs, 2);
      rs += __shfl_xor(rs, 4);
      rs += __shfl_xor(rs, 8);
      lrun[i] = lrun[i] * sc + rs;
      o[0][i] *= sc; o[1][i] *= sc; o[2][i] *= sc; o[3][i] *= sc;
      ps[(i + 8 * half) * 32 + m]      = (unsigned short)f2bf(p0);
      ps[(i + 8 * half) * 32 + 16 + m] = (unsigned short)f2bf(p1);
    }
    asm volatile("s_wait_dscnt 0" ::: "memory");

    // P as A-fragment (16x32 bf16) from LDS
    Frag pf;
#pragma unroll
    for (int j = 0; j < 8; ++j) {
      int k0 = 2 * j + (half ? 16 : 0);
      pf.u[j] = pstage[wv][(m * 32 + k0) >> 1];
    }

    // O += P * V : B[k,n] = V^T[ht*16+n][kb0+k], key-pairs contiguous in [H,T]
#pragma unroll
    for (int ht = 0; ht < 4; ++ht) {
      Frag vf;
#pragma unroll
      for (int j = 0; j < 8; ++j) {
        int kk = 2 * j + (half ? 16 : 0);
        vf.u[j] = vt_b[((ht * 16 + m) * TSEQ + kb0 + kk) >> 1];
      }
      o[ht] = __builtin_amdgcn_wmma_f32_16x16x32_bf16(
          false, pf.v, false, vf.v, (short)0, o[ht], false, false);
    }
  }

  // epilogue: divide by row sums, write fp32 output [B,T,H]
#pragma unroll
  for (int i = 0; i < 8; ++i) {
    float inv = 1.0f / lrun[i];
    int row = rowbase + i + 8 * half;
#pragma unroll
    for (int ht = 0; ht < 4; ++ht)
      out[row * HDIM + ht * 16 + m] = o[ht][i] * inv;
  }
}

// ---------------------------------------------------------------------------
extern "C" void kernel_launch(void* const* d_in, const int* in_sizes, int n_in,
                              void* d_out, int out_size, void* d_ws, size_t ws_size,
                              hipStream_t stream) {
  (void)in_sizes; (void)n_in; (void)out_size; (void)ws_size;
  const float* x  = (const float*)d_in[0];
  const float* Wk = (const float*)d_in[1];
  const float* Wq = (const float*)d_in[2];
  const float* Wv = (const float*)d_in[3];
  float* out = (float*)d_out;

  char* ws = (char*)d_ws;
  const size_t WT_BYTES = (size_t)CDIM * HDIM * 2;          // 96 KB each
  const size_t QK_BYTES = (size_t)BATCH * TSEQ * HDIM * 2;  // 2 MB each
  unsigned short* wqT = (unsigned short*)(ws);
  unsigned short* wkT = (unsigned short*)(ws + WT_BYTES);
  unsigned short* wvT = (unsigned short*)(ws + 2 * WT_BYTES);
  unsigned short* qbf = (unsigned short*)(ws + 3 * WT_BYTES);
  unsigned short* kbf = (unsigned short*)(ws + 3 * WT_BYTES + QK_BYTES);
  unsigned short* vtb = (unsigned short*)(ws + 3 * WT_BYTES + 2 * QK_BYTES);

  const int welems = CDIM * HDIM;
  const int wblocks = (welems + 255) / 256;
  wconvT_kernel<<<wblocks, 256, 0, stream>>>(Wq, wqT);
  wconvT_kernel<<<wblocks, 256, 0, stream>>>(Wk, wkT);
  wconvT_kernel<<<wblocks, 256, 0, stream>>>(Wv, wvT);

  const int tiles = BATCH * TSEQ / 16;   // 1024 waves
  qkv_proj_kernel<<<tiles / NW, NW * 32, 0, stream>>>(
      x, (const unsigned int*)wqT, (const unsigned int*)wkT,
      (const unsigned int*)wvT, qbf, kbf, vtb);

  flash_attn_kernel<<<tiles / NW, NW * 32, 0, stream>>>(
      (const unsigned int*)qbf, (const unsigned int*)kbf,
      (const unsigned int*)vtb, out);
}
